// DecoderTransformer_28905129902638
// MI455X (gfx1250) — matrix operand order repaired and checked by reading, not compiled, not run on hardware
//
#include <hip/hip_runtime.h>
#include <hip/hip_bf16.h>
#include <cstdint>

typedef __attribute__((ext_vector_type(16))) __bf16 v16bf;
typedef __attribute__((ext_vector_type(8)))  float  v8f;
typedef unsigned short u16;
typedef uint32_t u32;

namespace cfg {
constexpr int V = 32000, D = 1024, H = 16, L = 8, DH = 64, DF = 4096;
constexpr int B = 2, T = 1024, M = 64, S = M + T;   // S = 1088
constexpr int BT = B * T;                           // 2048
constexpr int BSQ = B * S;                          // 2176
constexpr float EPS = 1e-5f;
}

union FragAB { v16bf v; u32 u[8]; };

__device__ inline u16 f2bf(float x) {
  u32 u = __float_as_uint(x);
  u += 0x7FFFu + ((u >> 16) & 1u);     // round-to-nearest-even to bf16
  return (u16)(u >> 16);
}
__device__ inline float gelu_tanh(float x) {
  return 0.5f * x * (1.0f + tanhf(0.7978845608f * (x + 0.044715f * x * x * x)));
}
__device__ inline v8f v8f_zero() {
  v8f z = {0.f, 0.f, 0.f, 0.f, 0.f, 0.f, 0.f, 0.f};
  return z;
}
// Async DMA global -> LDS, 8 bytes per lane, tracked by ASYNCcnt.
__device__ inline void async_b64(unsigned lds_off, const void* gaddr) {
  asm volatile("global_load_async_to_lds_b64 %0, %1, off"
               :: "v"(lds_off), "v"((uint64_t)(uintptr_t)gaddr) : "memory");
}
__device__ inline void wait_async6() { asm volatile("s_wait_asynccnt 0x6" ::: "memory"); }
__device__ inline void wait_async0() { asm volatile("s_wait_asynccnt 0x0" ::: "memory"); }

// ---------------------------------------------------------------------------
// Weight prep: W[K,N] fp32 -> WT[N,K] bf16 (transposed so K is contiguous)
// ---------------------------------------------------------------------------
__global__ void __launch_bounds__(256) transpose_bf16(
    const float* __restrict__ W, u16* __restrict__ WT, int K, int N) {
  __shared__ float t[32][33];
  const int tid = threadIdx.x;
  const int tx = tid & 31, ty = tid >> 5;            // 8 rows per pass
  const int k0 = blockIdx.y * 32, n0 = blockIdx.x * 32;
#pragma unroll
  for (int i = 0; i < 4; ++i) {
    const int ky = ty + i * 8;
    t[ky][tx] = W[(size_t)(k0 + ky) * N + n0 + tx];
  }
  __syncthreads();
#pragma unroll
  for (int i = 0; i < 4; ++i) {
    const int ny = ty + i * 8;
    WT[(size_t)(n0 + ny) * K + k0 + tx] = f2bf(t[tx][ny]);
  }
}

// ---------------------------------------------------------------------------
// Embedding (residual stream h stays fp32)
// ---------------------------------------------------------------------------
__global__ void __launch_bounds__(256) embed_kernel(
    const int* __restrict__ x, const float* __restrict__ mem,
    const float* __restrict__ tok, const float* __restrict__ pos,
    float* __restrict__ hOut) {
  using namespace cfg;
  const size_t idx = (size_t)blockIdx.x * 256 + threadIdx.x;
  if (idx >= (size_t)BSQ * D) return;
  const int d = (int)(idx % D);
  const int rs = (int)(idx / D);
  const int b = rs / S, s = rs % S;
  float v;
  if (s < M) {
    v = mem[((size_t)b * M + s) * D + d];
  } else {
    const int t = s - M;
    v = tok[(size_t)x[b * T + t] * D + d] + pos[(size_t)t * D + d];
  }
  hOut[idx] = v;
}

// ---------------------------------------------------------------------------
// LayerNorm over D: fp32 in, bf16 out (feeds WMMA GEMMs directly)
// ---------------------------------------------------------------------------
__global__ void __launch_bounds__(256) ln_kernel(
    const float* __restrict__ in, u16* __restrict__ out,
    const float* __restrict__ sc, const float* __restrict__ bi,
    int Trows, int Sin, int rowOff) {
  using namespace cfg;
  __shared__ float r1[256], r2[256];
  const int tid = threadIdx.x;
  const int row = blockIdx.x;
  const int bb = row / Trows, tt = row % Trows;
  const float* src = in + ((size_t)bb * Sin + rowOff + tt) * D;
  u16* dst = out + (size_t)row * D;
  float sum = 0.f, sq = 0.f;
  for (int i = tid; i < D; i += 256) { const float v = src[i]; sum += v; sq += v * v; }
  r1[tid] = sum; r2[tid] = sq;
  __syncthreads();
  for (int s = 128; s > 0; s >>= 1) {
    if (tid < s) { r1[tid] += r1[tid + s]; r2[tid] += r2[tid + s]; }
    __syncthreads();
  }
  const float mean = r1[0] * (1.f / D);
  const float var  = r2[0] * (1.f / D) - mean * mean;
  const float rstd = rsqrtf(var + EPS);
  for (int i = tid; i < D; i += 256)
    dst[i] = f2bf((src[i] - mean) * rstd * sc[i] + bi[i]);
}

// ---------------------------------------------------------------------------
// WMMA GEMM: C[M,N] = epi(A[M,K]bf16 @ BT[N,K]bf16^T)
// Block 256 thr (8 wave32), tile 64x128, BK=32, double-buffered LDS filled by
// GLOBAL_LOAD_ASYNC_TO_LDS_B64 (ASYNCcnt), 4 WMMAs per wave per K-step.
// Wave w: m-half = w>>2 (32 rows), n-quarter = w&3 (32 cols) -> 2x2 tiles.
// ---------------------------------------------------------------------------
template <bool BIAS, bool GELU, bool RESID, bool OUTBF>
__global__ void __launch_bounds__(256) gemm_wmma(
    const u16* __restrict__ A, const u16* __restrict__ Bt,
    const float* __restrict__ bias, const float* __restrict__ Rm,
    void* __restrict__ Cout, int N, int K) {
  __shared__ u16 sA[2][64 * 36];    // [row][k], stride 36 (72B, b64-aligned)
  __shared__ u16 sB[2][128 * 36];   // [n][k],  stride 36
  const int tid  = threadIdx.x;
  const int lane = tid & 31, wv = tid >> 5;
  const int hl = lane >> 4, lm = lane & 15;
  const int nq = wv & 3;
  const int mh = wv >> 2;
  const int rowBase = blockIdx.y * 64;
  const int colBase = blockIdx.x * 128;

  v8f acc00 = v8f_zero(), acc01 = v8f_zero();
  v8f acc10 = v8f_zero(), acc11 = v8f_zero();

  auto stage = [&](int buf, int k0) {
    // A tile 64x32 bf16 = 512 x b64 chunks (2 per thread)
    for (int c = tid; c < 512; c += 256) {
      const int r = c >> 3, c8 = (c & 7) * 4;
      async_b64((unsigned)(uintptr_t)&sA[buf][r * 36 + c8],
                A + (size_t)(rowBase + r) * K + k0 + c8);
    }
    // B tile 128x32 bf16 = 1024 x b64 chunks (4 per thread)
    for (int c = tid; c < 1024; c += 256) {
      const int n = c >> 3, c8 = (c & 7) * 4;
      async_b64((unsigned)(uintptr_t)&sB[buf][n * 36 + c8],
                Bt + (size_t)(colBase + n) * K + k0 + c8);
    }
  };

  const int nk = K / 32;
  stage(0, 0);
  for (int t = 0; t < nk; ++t) {
    if (t + 1 < nk) { stage((t + 1) & 1, (t + 1) * 32); wait_async6(); }
    else            { wait_async0(); }
    __syncthreads();                       // tile t visible to all waves

    const u16* sAb = sA[t & 1];
    const u16* sBb = sB[t & 1];
    FragAB a0, a1, b0, b1;
#pragma unroll
    for (int j = 0; j < 8; ++j) {
      const int kb = ((j & 4) ? 16 : 0) + hl * 8 + (j & 3) * 2;   // A K-map
      a0.u[j] = *(const u32*)&sAb[(mh * 32 + lm) * 36 + kb];
      a1.u[j] = *(const u32*)&sAb[(mh * 32 + 16 + lm) * 36 + kb];
    }
#pragma unroll
    for (int j = 0; j < 8; ++j) {
      const int kk = ((j & 4) ? 16 : 0) + hl * 8 + (j & 3) * 2;   // B K-map
      b0.u[j] = *(const u32*)&sBb[(nq * 32 + lm) * 36 + kk];
      b1.u[j] = *(const u32*)&sBb[(nq * 32 + 16 + lm) * 36 + kk];
    }
    acc00 = __builtin_amdgcn_wmma_f32_16x16x32_bf16(false, a0.v, false, b0.v, (short)0, acc00, false, false);
    acc01 = __builtin_amdgcn_wmma_f32_16x16x32_bf16(false, a0.v, false, b1.v, (short)0, acc01, false, false);
    acc10 = __builtin_amdgcn_wmma_f32_16x16x32_bf16(false, a1.v, false, b0.v, (short)0, acc10, false, false);
    acc11 = __builtin_amdgcn_wmma_f32_16x16x32_bf16(false, a1.v, false, b1.v, (short)0, acc11, false, false);
    __syncthreads();                       // done reading before next DMA hits it
  }

  // epilogue: D layout row = base + r + 8*hl, col = base + (lane&15)
#pragma unroll
  for (int mt = 0; mt < 2; ++mt) {
#pragma unroll
    for (int nt = 0; nt < 2; ++nt) {
      const v8f acc = mt ? (nt ? acc11 : acc10) : (nt ? acc01 : acc00);
      const int col = colBase + nq * 32 + nt * 16 + lm;
#pragma unroll
      for (int r = 0; r < 8; ++r) {
        const int row = rowBase + mh * 32 + mt * 16 + hl * 8 + r;
        float vO = acc[r];
        if (BIAS)  vO += bias[col];
        if (GELU)  vO = gelu_tanh(vO);
        if (RESID) vO += Rm[(size_t)row * N + col];
        const size_t idx = (size_t)row * N + col;
        if (OUTBF) ((u16*)Cout)[idx] = f2bf(vO);
        else       ((float*)Cout)[idx] = vO;
      }
    }
  }
}

// ---------------------------------------------------------------------------
// Flash attention on bf16 Q/K/V: 4 waves, each owns 16 q-rows of one (b,head).
// K staged via async DMA; V via transposing LDS store; QK^T and P.V in WMMA.
// ---------------------------------------------------------------------------
__global__ void __launch_bounds__(128) attn_flash(
    const u16* __restrict__ Q, const u16* __restrict__ Kc,
    const u16* __restrict__ Vc, u16* __restrict__ O) {
  using namespace cfg;
  __shared__ u16 sK[32 * 68];      // [key][dh], stride 68 (136B, b64-aligned)
  __shared__ u16 sV[64 * 34];      // [dh][key], stride 34
  __shared__ u16 sP[4][16 * 34];   // per-wave P transpose area

  const int tid = threadIdx.x;
  const int w = tid >> 5, lane = tid & 31;
  const int hl = lane >> 4, lm = lane & 15;
  const int bh = blockIdx.y;
  const int b = bh / H, h = bh % H;
  const int qrow0 = blockIdx.x * 64 + w * 16;

  // Q fragments (A layout): direct bf16-pair (u32) loads, no conversion
  FragAB qf[2];
  {
    const u16* qp = Q + ((size_t)b * S + qrow0 + lm) * D + h * DH;
#pragma unroll
    for (int f = 0; f < 2; ++f)
#pragma unroll
      for (int j = 0; j < 8; ++j) {
        const int dh = f * 32 + ((j & 4) ? 16 : 0) + hl * 8 + (j & 3) * 2;
        qf[f].u[j] = *(const u32*)(qp + dh);
      }
  }

  v8f oacc[4];
#pragma unroll
  for (int t = 0; t < 4; ++t) oacc[t] = v8f_zero();
  float mrow[8], lrow[8];
#pragma unroll
  for (int r = 0; r < 8; ++r) { mrow[r] = -1e30f; lrow[r] = 0.f; }

  const int ktEnd = blockIdx.x * 2 + 2;        // causal early exit
  for (int kt = 0; kt < ktEnd; ++kt) {
    const int key0 = kt * 32;
    // K tile 32x64 bf16: 512 b64 chunks via async DMA (4 per thread)
    for (int c = tid; c < 512; c += 128) {
      const int kk = c >> 4, c8 = (c & 15) * 4;
      async_b64((unsigned)(uintptr_t)&sK[kk * 68 + c8],
                Kc + ((size_t)b * S + key0 + kk) * D + h * DH + c8);
    }
    // V tile transposed to [dh][key]: u32 load (2 dh), 2 u16 scatter stores
    for (int c = tid; c < 1024; c += 128) {
      const int kk = c >> 5, dd = (c & 31) * 2;
      const u32 wv2 = *(const u32*)(Vc + ((size_t)b * S + key0 + kk) * D + h * DH + dd);
      sV[dd * 34 + kk]       = (u16)wv2;
      sV[(dd + 1) * 34 + kk] = (u16)(wv2 >> 16);
    }
    wait_async0();
    __syncthreads();

    // scores: two 16-key subtiles, K-dim = DH = 64 -> 2 WMMAs each
    v8f sc[2];
#pragma unroll
    for (int s = 0; s < 2; ++s) {
      sc[s] = v8f_zero();
#pragma unroll
      for (int f = 0; f < 2; ++f) {
        FragAB kb;
#pragma unroll
        for (int j = 0; j < 8; ++j) {
          const int dh = f * 32 + ((j & 4) ? 16 : 0) + hl * 8 + (j & 3) * 2;
          kb.u[j] = *(const u32*)&sK[(s * 16 + lm) * 68 + dh];
        }
        sc[s] = __builtin_amdgcn_wmma_f32_16x16x32_bf16(false, qf[f].v, false, kb.v, (short)0, sc[s], false, false);
      }
      const int ki = key0 + s * 16 + lm;
#pragma unroll
      for (int r = 0; r < 8; ++r) {
        const int qi = qrow0 + hl * 8 + r;
        const float xv = sc[s][r] * 0.125f;     // 1/sqrt(DH)
        sc[s][r] = (ki <= qi) ? xv : -1e30f;
      }
    }

    // online softmax: row reductions within 16-lane halves
#pragma unroll
    for (int r = 0; r < 8; ++r) {
      float mt = fmaxf(sc[0][r], sc[1][r]);
#pragma unroll
      for (int off = 1; off < 16; off <<= 1)
        mt = fmaxf(mt, __shfl_xor(mt, off, 32));
      const float mnew = fmaxf(mrow[r], mt);
      const float scl = __expf(mrow[r] - mnew);
      const float p0 = __expf(sc[0][r] - mnew);
      const float p1 = __expf(sc[1][r] - mnew);
      sc[0][r] = p0; sc[1][r] = p1;
      float rs = p0 + p1;
#pragma unroll
      for (int off = 1; off < 16; off <<= 1)
        rs += __shfl_xor(rs, off, 32);
      lrow[r] = lrow[r] * scl + rs;
      mrow[r] = mnew;
#pragma unroll
      for (int t = 0; t < 4; ++t) oacc[t][r] *= scl;
    }

    // transpose P (D layout -> A layout) through per-wave LDS
#pragma unroll
    for (int s = 0; s < 2; ++s)
#pragma unroll
      for (int r = 0; r < 8; ++r)
        sP[w][(hl * 8 + r) * 34 + s * 16 + lm] = f2bf(sc[s][r]);

    FragAB pf;
#pragma unroll
    for (int j = 0; j < 8; ++j) {
      const int key = ((j & 4) ? 16 : 0) + hl * 8 + (j & 3) * 2;
      pf.u[j] = *(const u32*)&sP[w][lm * 34 + key];
    }

    // P(16x32) @ V(32x64): 4 dh tiles
#pragma unroll
    for (int t = 0; t < 4; ++t) {
      FragAB vb;
#pragma unroll
      for (int j = 0; j < 8; ++j) {
        const int key = ((j & 4) ? 16 : 0) + hl * 8 + (j & 3) * 2;
        vb.u[j] = *(const u32*)&sV[(t * 16 + lm) * 34 + key];
      }
      oacc[t] = __builtin_amdgcn_wmma_f32_16x16x32_bf16(false, pf.v, false, vb.v, (short)0, oacc[t], false, false);
    }
    __syncthreads();
  }

#pragma unroll
  for (int t = 0; t < 4; ++t)
#pragma unroll
    for (int r = 0; r < 8; ++r) {
      const int row = qrow0 + hl * 8 + r;
      O[((size_t)b * S + row) * D + h * DH + t * 16 + lm] = f2bf(oacc[t][r] / lrow[r]);
    }
}

// ---------------------------------------------------------------------------
// Cross-entropy: per-row logsumexp then deterministic reduction
// ---------------------------------------------------------------------------
__global__ void __launch_bounds__(256) loss_row_kernel(
    const float* __restrict__ logits, const int* __restrict__ labels,
    const int* __restrict__ ignore, float* __restrict__ nll, float* __restrict__ cnt) {
  using namespace cfg;
  __shared__ float red[256];
  const int tid = threadIdx.x;
  const int row = blockIdx.x;
  const float* lg = logits + (size_t)row * V;
  float mx = -3.0e38f;
  for (int i = tid; i < V; i += 256) mx = fmaxf(mx, lg[i]);
  red[tid] = mx; __syncthreads();
  for (int s = 128; s > 0; s >>= 1) {
    if (tid < s) red[tid] = fmaxf(red[tid], red[tid + s]);
    __syncthreads();
  }
  mx = red[0];
  __syncthreads();
  float se = 0.f;
  for (int i = tid; i < V; i += 256) se += __expf(lg[i] - mx);
  red[tid] = se; __syncthreads();
  for (int s = 128; s > 0; s >>= 1) {
    if (tid < s) red[tid] += red[tid + s];
    __syncthreads();
  }
  if (tid == 0) {
    const float lse = mx + logf(red[0]);
    const int lbl = labels[row];
    const bool valid = (lbl != ignore[0]);
    const int safe = valid ? lbl : 0;
    nll[row] = valid ? (lse - lg[safe]) : 0.f;
    cnt[row] = valid ? 1.f : 0.f;
  }
}

__global__ void __launch_bounds__(256) loss_final_kernel(
    const float* __restrict__ nll, const float* __restrict__ cnt,
    float* __restrict__ out) {
  using namespace cfg;
  __shared__ float r1[256], r2[256];
  const int tid = threadIdx.x;
  float a = 0.f, c = 0.f;
  for (int i = tid; i < BT; i += 256) { a += nll[i]; c += cnt[i]; }
  r1[tid] = a; r2[tid] = c; __syncthreads();
  for (int s = 128; s > 0; s >>= 1) {
    if (tid < s) { r1[tid] += r1[tid + s]; r2[tid] += r2[tid + s]; }
    __syncthreads();
  }
  if (tid == 0) out[0] = r1[0] / fmaxf(r2[0], 1.f);
}

// ---------------------------------------------------------------------------
extern "C" void kernel_launch(void* const* d_in, const int* in_sizes, int n_in,
                              void* d_out, int out_size, void* d_ws, size_t ws_size,
                              hipStream_t stream) {
  using namespace cfg;
  const int*   x    = (const int*)  d_in[0];
  const float* mem  = (const float*)d_in[1];
  const int*   lbl  = (const int*)  d_in[2];
  const int*   ign  = (const int*)  d_in[3];
  const float* tok  = (const float*)d_in[4];
  const float* pos  = (const float*)d_in[5];
  const float* ln1s = (const float*)d_in[6];
  const float* ln1b = (const float*)d_in[7];
  const float* Wq   = (const float*)d_in[8];
  const float* Wk   = (const float*)d_in[9];
  const float* Wv   = (const float*)d_in[10];
  const float* Wo   = (const float*)d_in[11];
  const float* ln2s = (const float*)d_in[12];
  const float* ln2b = (const float*)d_in[13];
  const float* W1   = (const float*)d_in[14];
  const float* b1   = (const float*)d_in[15];
  const float* W2   = (const float*)d_in[16];
  const float* b2   = (const float*)d_in[17];
  const float* lnfs = (const float*)d_in[18];
  const float* lnfb = (const float*)d_in[19];
  const float* Wlm  = (const float*)d_in[20];
  const float* blm  = (const float*)d_in[21];

  // ---- workspace bump allocator ----
  char* wp = (char*)d_ws;
  auto alloc = [&](size_t bytes) -> void* {
    void* p = (void*)wp;
    wp += (bytes + 63) & ~(size_t)63;
    return p;
  };
  float* h    = (float*)alloc((size_t)BSQ * D * 4);
  u16*   a    = (u16*)  alloc((size_t)BSQ * D * 2);
  u16*   q    = (u16*)  alloc((size_t)BSQ * D * 2);
  u16*   k    = (u16*)  alloc((size_t)BSQ * D * 2);
  u16*   v    = (u16*)  alloc((size_t)BSQ * D * 2);
  u16*   o    = (u16*)  alloc((size_t)BSQ * D * 2);
  u16*   f1   = (u16*)  alloc((size_t)BSQ * DF * 2);
  u16*   lnf  = (u16*)  alloc((size_t)BT * D * 2);
  float* nll  = (float*)alloc((size_t)BT * 4);
  float* cnt  = (float*)alloc((size_t)BT * 4);
  u16* WqT  = (u16*)alloc((size_t)L * D * D * 2);
  u16* WkT  = (u16*)alloc((size_t)L * D * D * 2);
  u16* WvT  = (u16*)alloc((size_t)L * D * D * 2);
  u16* WoT  = (u16*)alloc((size_t)L * D * D * 2);
  u16* W1T  = (u16*)alloc((size_t)L * D * DF * 2);
  u16* W2T  = (u16*)alloc((size_t)L * DF * D * 2);
  u16* WlmT = (u16*)alloc((size_t)D * V * 2);

  float* logits = (float*)d_out;
  float* loss   = logits + (size_t)BT * V;

  // ---- one-time (per launch) weight conversion: fp32 [K,N] -> bf16 [N,K] ----
  const dim3 gDD(D / 32, D / 32);
  for (int i = 0; i < L; ++i) {
    const size_t oDD = (size_t)i * D * D;
    transpose_bf16<<<gDD, 256, 0, stream>>>(Wq + oDD, WqT + oDD, D, D);
    transpose_bf16<<<gDD, 256, 0, stream>>>(Wk + oDD, WkT + oDD, D, D);
    transpose_bf16<<<gDD, 256, 0, stream>>>(Wv + oDD, WvT + oDD, D, D);
    transpose_bf16<<<gDD, 256, 0, stream>>>(Wo + oDD, WoT + oDD, D, D);
    transpose_bf16<<<dim3(DF / 32, D / 32), 256, 0, stream>>>(W1 + (size_t)i * D * DF, W1T + (size_t)i * D * DF, D, DF);
    transpose_bf16<<<dim3(D / 32, DF / 32), 256, 0, stream>>>(W2 + (size_t)i * DF * D, W2T + (size_t)i * DF * D, DF, D);
  }
  transpose_bf16<<<dim3(V / 32, D / 32), 256, 0, stream>>>(Wlm, WlmT, D, V);

  embed_kernel<<<(unsigned)(((size_t)BSQ * D + 255) / 256), 256, 0, stream>>>(x, mem, tok, pos, h);

  const dim3 gQKV(D / 128, BSQ / 64);
  const dim3 gFF1(DF / 128, BSQ / 64);
  const dim3 gAtt(S / 64, B * H);

  for (int i = 0; i < L; ++i) {
    ln_kernel<<<BSQ, 256, 0, stream>>>(h, a, ln1s + (size_t)i * D, ln1b + (size_t)i * D, S, S, 0);
    gemm_wmma<false, false, false, true ><<<gQKV, 256, 0, stream>>>(a, WqT + (size_t)i * D * D, nullptr, nullptr, q, D, D);
    gemm_wmma<false, false, false, true ><<<gQKV, 256, 0, stream>>>(a, WkT + (size_t)i * D * D, nullptr, nullptr, k, D, D);
    gemm_wmma<false, false, false, true ><<<gQKV, 256, 0, stream>>>(a, WvT + (size_t)i * D * D, nullptr, nullptr, v, D, D);
    attn_flash<<<gAtt, 128, 0, stream>>>(q, k, v, o);
    gemm_wmma<false, false, true , false><<<gQKV, 256, 0, stream>>>(o, WoT + (size_t)i * D * D, nullptr, h, h, D, D);
    ln_kernel<<<BSQ, 256, 0, stream>>>(h, a, ln2s + (size_t)i * D, ln2b + (size_t)i * D, S, S, 0);
    gemm_wmma<true , true , false, true ><<<gFF1, 256, 0, stream>>>(a, W1T + (size_t)i * D * DF, b1 + (size_t)i * DF, nullptr, f1, DF, D);
    gemm_wmma<true , false, true , false><<<gQKV, 256, 0, stream>>>(f1, W2T + (size_t)i * DF * D, b2 + (size_t)i * D, h, h, D, DF);
  }
  ln_kernel<<<BT, 256, 0, stream>>>(h, lnf, lnfs, lnfb, T, S, M);
  gemm_wmma<true, false, false, false><<<dim3(V / 128, BT / 64), 256, 0, stream>>>(lnf, WlmT, blm, nullptr, logits, V, D);
  loss_row_kernel<<<BT, 256, 0, stream>>>(logits, lbl, ign, nll, cnt);
  loss_final_kernel<<<1, 256, 0, stream>>>(nll, cnt, loss);

  (void)in_sizes; (void)n_in; (void)out_size; (void)ws_size;
}
